// VisionAttention_48421461295263
// MI455X (gfx1250) — matrix operand order repaired and checked by reading, not compiled, not run on hardware
//
#include <hip/hip_runtime.h>
#include <hip/hip_bf16.h>

typedef __attribute__((ext_vector_type(16))) _Float16 v16h;
typedef __attribute__((ext_vector_type(8)))  _Float16 v8h;
typedef __attribute__((ext_vector_type(8)))  float    v8f;

// ---------------------------------------------------------------------------
// WMMA helpers (CDNA5 gfx1250, wave32)
// ---------------------------------------------------------------------------
__device__ __forceinline__ v8f wmma_f16(v16h a, v16h b, v8f c) {
  return __builtin_amdgcn_wmma_f32_16x16x32_f16(
      false, a, false, b, (short)0, c, false, false);
}

__device__ __forceinline__ v8f vzero8f() {
  v8f r;
#pragma unroll
  for (int i = 0; i < 8; ++i) r[i] = 0.0f;
  return r;
}

// A fragment, 16x32 f16: lane<16 -> row M=lane, K {0..7,16..23};
// lane>=16 -> row M=lane-16, K {8..15,24..31}. Row-major source, ld halves.
__device__ __forceinline__ v16h load_fragA(const _Float16* p, int ld, int r0,
                                           int k0, int lane) {
  int r  = r0 + (lane & 15);
  int ko = k0 + ((lane >> 4) << 3);
  const _Float16* q = p + (size_t)r * ld + ko;
  v8h x = *(const v8h*)(q);
  v8h y = *(const v8h*)(q + 16);
  v16h out;
#pragma unroll
  for (int i = 0; i < 8; ++i) { out[i] = x[i]; out[i + 8] = y[i]; }
  return out;
}

// B fragment, 32x16 f16: lane<16 -> col N=lane, K {0..7,16..23}; lane>=16 -> K {8..15,24..31}.
// Source stored TRANSPOSED [N][K], ld halves.
__device__ __forceinline__ v16h load_fragB(const _Float16* p, int ld, int n0,
                                           int k0, int lane) {
  int n  = n0 + (lane & 15);
  int ko = k0 + ((lane >> 4) << 3);
  const _Float16* q = p + (size_t)n * ld + ko;
  v8h x = *(const v8h*)(q);
  v8h y = *(const v8h*)(q + 16);
  v16h out;
#pragma unroll
  for (int i = 0; i < 8; ++i) { out[i] = x[i]; out[i + 8] = y[i]; }
  return out;
}

// Async global -> LDS copy (ASYNCcnt path), per-lane 16B.
__device__ __forceinline__ void async_b128(unsigned ldsoff, const void* g) {
  asm volatile("global_load_async_to_lds_b128 %0, %1, off"
               :: "v"(ldsoff), "v"(g) : "memory");
}
__device__ __forceinline__ void wait_async8() {
  asm volatile("s_wait_asynccnt 8" ::: "memory");
}
__device__ __forceinline__ void wait_async0() {
  asm volatile("s_wait_asynccnt 0" ::: "memory");
}

// ---------------------------------------------------------------------------
// Pack kernels: fp32 -> f16 hi/lo split (compensated f16x3 GEMM inputs)
// ---------------------------------------------------------------------------
__global__ void pack_split(const float* __restrict__ x, _Float16* __restrict__ hi,
                           _Float16* __restrict__ lo, int n) {
  int i = blockIdx.x * blockDim.x + threadIdx.x;
  if (i >= n) return;
  float v = x[i];
  _Float16 h = (_Float16)v;
  hi[i] = h;
  lo[i] = (_Float16)(v - (float)h);
}

// weights [Kd][Nn] fp32 -> transposed [Nn][Kd] f16 hi/lo
__global__ void pack_split_t(const float* __restrict__ x, _Float16* __restrict__ hi,
                             _Float16* __restrict__ lo, int Kd, int Nn) {
  int i = blockIdx.x * blockDim.x + threadIdx.x;
  if (i >= Kd * Nn) return;
  int k = i % Kd;
  int n = i / Kd;
  float v = x[(size_t)k * Nn + n];
  _Float16 h = (_Float16)v;
  hi[i] = h;
  lo[i] = (_Float16)(v - (float)h);
}

// ---------------------------------------------------------------------------
// Compensated f16x3 GEMM with async double-buffered LDS staging.
// C[M][Nn] = A[M][Kd] * B[Kd][Nn] + bias; A hi/lo row-major, B hi/lo [Nn][Kd].
// Block = 256 threads = 8 waves (4x2), WG tile 128x128, wave tile 32x64.
// LDS: 2 stages x 4 panels (Ah,Al,Bh,Bl) x 128x32 f16 = 64 KB.
// ---------------------------------------------------------------------------
__global__ void __launch_bounds__(256) gemm_f16x3(
    const _Float16* __restrict__ Ah, const _Float16* __restrict__ Al,
    const _Float16* __restrict__ Bth, const _Float16* __restrict__ Btl,
    const float* __restrict__ bias, float* __restrict__ C,
    int M, int Nn, int Kd) {
  __shared__ _Float16 sm[2][4][128 * 32];
  int tid  = threadIdx.x;
  int lane = tid & 31;
  int w    = tid >> 5;
  int wr   = w >> 1;   // 0..3
  int wc   = w & 1;    // 0..1
  int rowblk = blockIdx.x * 128;
  int colblk = blockIdx.y * 128;
  unsigned smbase = (unsigned)(uintptr_t)(&sm[0][0][0]);

  const _Float16* srcs[4] = {Ah, Al, Bth, Btl};

  // stage one K-slice (k0) of all 4 panels into buffer `buf`
  auto stage = [&](int buf, int k0) {
#pragma unroll
    for (int t = 0; t < 4; ++t) {
      const _Float16* src = srcs[t];
      int base = (t < 2) ? rowblk : colblk;
#pragma unroll
      for (int c2 = 0; c2 < 2; ++c2) {
        int c    = tid * 2 + c2;   // 0..511 chunks of 8 halves
        int r    = c >> 2;         // 0..127 panel row (A-row or B-col)
        int part = c & 3;          // 16B chunk within 64B row
        const void* g = src + (size_t)(base + r) * Kd + k0 + part * 8;
        unsigned lo = smbase + (unsigned)((buf * 4 + t) * (128 * 32) + c * 8) * 2u;
        async_b128(lo, g);
      }
    }
  };

  v8f acc[2][4];
#pragma unroll
  for (int i = 0; i < 2; ++i)
#pragma unroll
    for (int j = 0; j < 4; ++j) acc[i][j] = vzero8f();

  int nk = Kd >> 5;  // K-steps of 32
  stage(0, 0);
  for (int ki = 0; ki < nk; ++ki) {
    int buf = ki & 1;
    if (ki + 1 < nk) {
      stage(buf ^ 1, (ki + 1) << 5);
      wait_async8();   // in-order: current stage's 8 loads have landed
    } else {
      wait_async0();
    }
    __syncthreads();

    const _Float16* As  = &sm[buf][0][0];
    const _Float16* Als = &sm[buf][1][0];
    const _Float16* Bs  = &sm[buf][2][0];
    const _Float16* Bls = &sm[buf][3][0];

    v16h ah[2], al[2], bh[4], bl[4];
#pragma unroll
    for (int i = 0; i < 2; ++i) {
      ah[i] = load_fragA(As,  32, wr * 32 + i * 16, 0, lane);
      al[i] = load_fragA(Als, 32, wr * 32 + i * 16, 0, lane);
    }
#pragma unroll
    for (int j = 0; j < 4; ++j) {
      bh[j] = load_fragB(Bs,  32, wc * 64 + j * 16, 0, lane);
      bl[j] = load_fragB(Bls, 32, wc * 64 + j * 16, 0, lane);
    }
#pragma unroll
    for (int i = 0; i < 2; ++i)
#pragma unroll
      for (int j = 0; j < 4; ++j) {
        acc[i][j] = wmma_f16(ah[i], bh[j], acc[i][j]);  // hi*hi
        acc[i][j] = wmma_f16(ah[i], bl[j], acc[i][j]);  // hi*lo
        acc[i][j] = wmma_f16(al[i], bh[j], acc[i][j]);  // lo*hi
      }
    __syncthreads();  // protect buf before it is refilled next iteration
  }

  int rr = (lane >> 4) << 3;
  int cc = lane & 15;
#pragma unroll
  for (int i = 0; i < 2; ++i)
#pragma unroll
    for (int j = 0; j < 4; ++j) {
      int col = colblk + wc * 64 + j * 16 + cc;
      float b = bias[col];
#pragma unroll
      for (int r = 0; r < 8; ++r) {
        int row = rowblk + wr * 32 + i * 16 + rr + r;
        C[(size_t)row * Nn + col] = acc[i][j][r] + b;
      }
    }
}

// ---------------------------------------------------------------------------
// RoPE + repack: qkv f32 [N][3][16][80] -> Qh/Kh [16][4096][96] f16 (pad 0),
// Vt [16][80][4096] f16 (transposed for B-fragment loads).
// ---------------------------------------------------------------------------
__global__ void rope_pack(const float* __restrict__ qkv,
                          const float* __restrict__ cosNK,
                          const float* __restrict__ sinNK,
                          _Float16* __restrict__ Qh, _Float16* __restrict__ Kh,
                          _Float16* __restrict__ Vt) {
  int idx = blockIdx.x * blockDim.x + threadIdx.x;
  if (idx >= 4096 * 16 * 96) return;
  int k  = idx % 96;
  int th = idx / 96;
  int h  = th % 16;
  int t  = th / 16;
  size_t qdst = ((size_t)h * 4096 + t) * 96 + k;
  if (k >= 80) {
    Qh[qdst] = (_Float16)0.0f;
    Kh[qdst] = (_Float16)0.0f;
    return;
  }
  size_t b = (size_t)t * 3840 + h * 80 + k;
  float q  = qkv[b];
  float kk = qkv[b + 1280];
  float v  = qkv[b + 2560];
  float c  = cosNK[t * 80 + k];
  float sn = sinNK[t * 80 + k];
  int   kro = (k < 40) ? (k + 40) : (k - 40);
  float sgn = (k < 40) ? -1.0f : 1.0f;
  size_t br = (size_t)t * 3840 + h * 80 + kro;
  float qr = sgn * qkv[br];
  float kr = sgn * qkv[br + 1280];
  Qh[qdst] = (_Float16)(q * c + qr * sn);
  Kh[qdst] = (_Float16)(kk * c + kr * sn);
  Vt[((size_t)h * 80 + k) * 4096 + t] = (_Float16)v;
}

// ---------------------------------------------------------------------------
// Attention: block-diagonal (16 segments x 256 tokens), 16 heads.
// Grid (4 rowblocks, 16 segs, 16 heads), 128 threads = 4 waves.
// ---------------------------------------------------------------------------
__global__ void __launch_bounds__(128) attn_kernel(
    const _Float16* __restrict__ Qh, const _Float16* __restrict__ Kh,
    const _Float16* __restrict__ Vt, float* __restrict__ O) {
  __shared__ _Float16 Plds[4][16][256];
  int lane = threadIdx.x & 31;
  int w    = threadIdx.x >> 5;
  int h    = blockIdx.z;
  int seg  = blockIdx.y;
  int row0 = seg * 256 + blockIdx.x * 64 + w * 16;

  const _Float16* Qb = Qh + (size_t)h * 4096 * 96;
  const _Float16* Kb = Kh + (size_t)h * 4096 * 96;
  const _Float16* Vb = Vt + (size_t)h * 80 * 4096;

  v16h qa[3];
#pragma unroll
  for (int ks = 0; ks < 3; ++ks) qa[ks] = load_fragA(Qb, 96, row0, ks * 32, lane);

  v8f s[16];
#pragma unroll
  for (int ct = 0; ct < 16; ++ct) {
    v8f a = vzero8f();
    int col0 = seg * 256 + ct * 16;
#pragma unroll
    for (int ks = 0; ks < 3; ++ks) {
      v16h kb = load_fragB(Kb, 96, col0, ks * 32, lane);
      a = wmma_f16(qa[ks], kb, a);
    }
    s[ct] = a;
  }

  const float scale = 0.11180339887498948f;  // 80^-0.5
  float rmax[8], rsum[8];
#pragma unroll
  for (int r = 0; r < 8; ++r) {
    float m = -3.0e38f;
#pragma unroll
    for (int ct = 0; ct < 16; ++ct) m = fmaxf(m, s[ct][r]);
    for (int d = 1; d < 16; d <<= 1) m = fmaxf(m, __shfl_xor(m, d, 32));
    rmax[r] = m;
    rsum[r] = 0.0f;
  }
#pragma unroll
  for (int ct = 0; ct < 16; ++ct)
#pragma unroll
    for (int r = 0; r < 8; ++r) {
      float e = __expf((s[ct][r] - rmax[r]) * scale);
      s[ct][r] = e;
      rsum[r] += e;
    }
#pragma unroll
  for (int r = 0; r < 8; ++r) {
    float t = rsum[r];
    for (int d = 1; d < 16; d <<= 1) t += __shfl_xor(t, d, 32);
    rsum[r] = 1.0f / t;
  }

  int rr = (lane >> 4) << 3;
  int cc = lane & 15;
#pragma unroll
  for (int ct = 0; ct < 16; ++ct)
#pragma unroll
    for (int r = 0; r < 8; ++r)
      Plds[w][rr + r][ct * 16 + cc] = (_Float16)(s[ct][r] * rsum[r]);
  __syncthreads();

  v8f o[5];
#pragma unroll
  for (int f = 0; f < 5; ++f) o[f] = vzero8f();
  for (int ks = 0; ks < 8; ++ks) {
    v16h pa = load_fragA(&Plds[w][0][0], 256, 0, ks * 32, lane);
#pragma unroll
    for (int f = 0; f < 5; ++f) {
      v16h vb = load_fragB(Vb, 4096, f * 16, seg * 256 + ks * 32, lane);
      o[f] = wmma_f16(pa, vb, o[f]);
    }
  }
#pragma unroll
  for (int f = 0; f < 5; ++f)
#pragma unroll
    for (int r = 0; r < 8; ++r) {
      int row = row0 + rr + r;
      O[(size_t)row * 1280 + h * 80 + f * 16 + cc] = o[f][r];
    }
}

// ---------------------------------------------------------------------------
// Launch
// ---------------------------------------------------------------------------
extern "C" void kernel_launch(void* const* d_in, const int* in_sizes, int n_in,
                              void* d_out, int out_size, void* d_ws, size_t ws_size,
                              hipStream_t stream) {
  (void)in_sizes; (void)n_in; (void)out_size; (void)ws_size;
  const float* hidden = (const float*)d_in[0];
  const float* cosNK = (const float*)d_in[2];
  const float* sinNK = (const float*)d_in[3];
  const float* qkvW  = (const float*)d_in[4];
  const float* qkvB  = (const float*)d_in[5];
  const float* projW = (const float*)d_in[6];
  const float* projB = (const float*)d_in[7];

  char* ws = (char*)d_ws;
  size_t off = 0;
  auto take = [&](size_t bytes) {
    char* p = ws + off;
    off += (bytes + 255) & ~(size_t)255;
    return p;
  };
  float*    bufA = (float*)take((size_t)4096 * 3840 * 4);
  _Float16* hidH = (_Float16*)take((size_t)4096 * 1280 * 2);
  _Float16* hidL = (_Float16*)take((size_t)4096 * 1280 * 2);
  _Float16* wqH  = (_Float16*)take((size_t)3840 * 1280 * 2);
  _Float16* wqL  = (_Float16*)take((size_t)3840 * 1280 * 2);
  _Float16* wpH  = (_Float16*)take((size_t)1280 * 1280 * 2);
  _Float16* wpL  = (_Float16*)take((size_t)1280 * 1280 * 2);
  _Float16* Qhp  = (_Float16*)take((size_t)16 * 4096 * 96 * 2);
  _Float16* Khp  = (_Float16*)take((size_t)16 * 4096 * 96 * 2);
  _Float16* Vtp  = (_Float16*)take((size_t)16 * 80 * 4096 * 2);

  pack_split<<<(4096 * 1280 + 255) / 256, 256, 0, stream>>>(hidden, hidH, hidL, 4096 * 1280);
  pack_split_t<<<(1280 * 3840 + 255) / 256, 256, 0, stream>>>(qkvW, wqH, wqL, 1280, 3840);
  pack_split_t<<<(1280 * 1280 + 255) / 256, 256, 0, stream>>>(projW, wpH, wpL, 1280, 1280);

  gemm_f16x3<<<dim3(32, 30), 256, 0, stream>>>(hidH, hidL, wqH, wqL, qkvB, bufA,
                                               4096, 3840, 1280);
  rope_pack<<<(4096 * 16 * 96 + 255) / 256, 256, 0, stream>>>(bufA, cosNK, sinNK,
                                                              Qhp, Khp, Vtp);
  attn_kernel<<<dim3(4, 16, 16), 128, 0, stream>>>(Qhp, Khp, Vtp, bufA);
  pack_split<<<(4096 * 1280 + 255) / 256, 256, 0, stream>>>(bufA, hidH, hidL, 4096 * 1280);
  gemm_f16x3<<<dim3(32, 10), 256, 0, stream>>>(hidH, hidL, wpH, wpL, projB,
                                               (float*)d_out, 4096, 1280, 1280);
}
